// SelfAttnEncoder_13340168421488
// MI455X (gfx1250) — matrix operand order repaired and checked by reading, not compiled, not run on hardware
//
#include <hip/hip_runtime.h>
#include <hip/hip_bf16.h>

#define B_   2
#define S_   2048
#define D_   1024
#define H_   16
#define DK_  64
#define OUT_ 1024

typedef __attribute__((ext_vector_type(16))) _Float16 v16h;
typedef __attribute__((ext_vector_type(8)))  _Float16 v8h;
typedef __attribute__((ext_vector_type(8)))  float    v8f;

__device__ __forceinline__ v8f wmma_f16(v16h a, v16h b, v8f c) {
  // D = A(16x32 f16) * B(32x16 f16) + C(16x16 f32)
  return __builtin_amdgcn_wmma_f32_16x16x32_f16(
      /*neg_a=*/false, a, /*neg_b=*/false, b,
      /*c_mod=*/(short)0, c, /*reuse_a=*/false, /*reuse_b=*/false);
}

// ---- A-operand loader (16xK tile): M = lane%16, K contiguous 16 per lane-half ----
__device__ __forceinline__ v16h ldA_f32(const float* base, int stride, int lane) {
  const int m = lane & 15, lh = (lane >> 4) & 1;
  const float* r0 = base + (size_t)m * stride + lh * 16;
  const float* r1 = r0 + 8;
  float4 a0 = *(const float4*)(r0);
  float4 a1 = *(const float4*)(r0 + 4);
  float4 b0 = *(const float4*)(r1);
  float4 b1 = *(const float4*)(r1 + 4);
  v16h v;
  v[0]=(_Float16)a0.x;  v[1]=(_Float16)a0.y;  v[2]=(_Float16)a0.z;  v[3]=(_Float16)a0.w;
  v[4]=(_Float16)a1.x;  v[5]=(_Float16)a1.y;  v[6]=(_Float16)a1.z;  v[7]=(_Float16)a1.w;
  v[8]=(_Float16)b0.x;  v[9]=(_Float16)b0.y;  v[10]=(_Float16)b0.z; v[11]=(_Float16)b0.w;
  v[12]=(_Float16)b1.x; v[13]=(_Float16)b1.y; v[14]=(_Float16)b1.z; v[15]=(_Float16)b1.w;
  return v;
}

__device__ __forceinline__ v16h ldA_lds(const _Float16* base, int stride, int lane) {
  const int m = lane & 15, lh = (lane >> 4) & 1;
  const _Float16* r0 = base + m * stride + lh * 16;
  v8h lo = *(const v8h*)(r0);
  v8h hi = *(const v8h*)(r0 + 8);
  return __builtin_shufflevector(lo, hi, 0,1,2,3,4,5,6,7,8,9,10,11,12,13,14,15);
}

// ---- B-operand loader (Kx16 tile): N = lane%16, K contiguous 16 per lane-half ----
// Source is stored N-major (row = N, K contiguous within row).
__device__ __forceinline__ v16h ldB_f32(const float* base, int stride, int lane) {
  const int n = lane & 15, lh = (lane >> 4) & 1;
  const float* r0 = base + (size_t)n * stride + lh * 16;
  float4 a0 = *(const float4*)(r0);
  float4 a1 = *(const float4*)(r0 + 4);
  float4 a2 = *(const float4*)(r0 + 8);
  float4 a3 = *(const float4*)(r0 + 12);
  v16h v;
  v[0]=(_Float16)a0.x;  v[1]=(_Float16)a0.y;  v[2]=(_Float16)a0.z;  v[3]=(_Float16)a0.w;
  v[4]=(_Float16)a1.x;  v[5]=(_Float16)a1.y;  v[6]=(_Float16)a1.z;  v[7]=(_Float16)a1.w;
  v[8]=(_Float16)a2.x;  v[9]=(_Float16)a2.y;  v[10]=(_Float16)a2.z; v[11]=(_Float16)a2.w;
  v[12]=(_Float16)a3.x; v[13]=(_Float16)a3.y; v[14]=(_Float16)a3.z; v[15]=(_Float16)a3.w;
  return v;
}

__device__ __forceinline__ v16h ldB_lds(const _Float16* base, int stride, int lane) {
  const int n = lane & 15, lh = (lane >> 4) & 1;
  const _Float16* r0 = base + n * stride + lh * 16;
  v8h lo = *(const v8h*)(r0);
  v8h hi = *(const v8h*)(r0 + 8);
  return __builtin_shufflevector(lo, hi, 0,1,2,3,4,5,6,7,8,9,10,11,12,13,14,15);
}

// =====================================================================
// Kernel 1: flash attention. Block = 256 thr (8 waves) -> (b, head, 128 q rows).
// Wave owns a 16-row q tile; block stages shared 32-key K/V tiles in LDS.
// =====================================================================
__global__ __launch_bounds__(256) void attn_flash_kernel(
    const float* __restrict__ Q, const float* __restrict__ K,
    const float* __restrict__ V, const float* __restrict__ mask,
    float* __restrict__ X) {
  __shared__ __align__(16) _Float16 kstage[32][64];      // [key][dk]     (B for QK^T)
  __shared__ __align__(16) _Float16 vstage[64][32];      // [dk][key]     (B for P*V)
  __shared__ __align__(16) _Float16 pstage[8][16][32];   // per-wave P transpose staging

  const int tid  = threadIdx.x;
  const int wave = tid >> 5, lane = tid & 31;
  const int m0 = lane & 15, lh = (lane >> 4) & 1;
  const int qc = blockIdx.x & 15;          // S/128 = 16 chunks
  const int hd = (blockIdx.x >> 4) & 15;   // head
  const int b  = blockIdx.x >> 8;          // batch
  const int qbase = qc * 128 + wave * 16;
  const size_t bs = (size_t)b * S_;

  // Q fragments for the whole K loop (f32 -> f16)
  const float* Qb = Q + (bs + qbase) * D_ + hd * DK_;
  v16h qa0 = ldA_f32(Qb, D_, lane);        // dk 0..31
  v16h qa1 = ldA_f32(Qb + 32, D_, lane);   // dk 32..63

  v8f o0 = {}, o1 = {}, o2 = {}, o3 = {};  // 16x64 output accumulator (D layout)
  float mrow[8], lrow[8];
#pragma unroll
  for (int r = 0; r < 8; ++r) { mrow[r] = -1e30f; lrow[r] = 0.f; }

  // cooperative staging roles: thread -> (key row, 8-wide dk segment)
  const int key = tid >> 3, seg = tid & 7;
  const float* Kg = K + (bs + key) * D_ + hd * DK_ + seg * 8;
  const float* Vg = V + (bs + key) * D_ + hd * DK_ + seg * 8;
  const float* mk = mask + bs;

  for (int kb = 0; kb < S_; kb += 32) {
    // ---- stage K (row-major f16) and V (transposed f16) into LDS ----
    {
      const float* kp = Kg + (size_t)kb * D_;
      float4 k0 = *(const float4*)kp, k1 = *(const float4*)(kp + 4);
      v8h kh;
      kh[0]=(_Float16)k0.x; kh[1]=(_Float16)k0.y; kh[2]=(_Float16)k0.z; kh[3]=(_Float16)k0.w;
      kh[4]=(_Float16)k1.x; kh[5]=(_Float16)k1.y; kh[6]=(_Float16)k1.z; kh[7]=(_Float16)k1.w;
      *(v8h*)&kstage[key][seg * 8] = kh;

      const float* vp = Vg + (size_t)kb * D_;
      float4 w0 = *(const float4*)vp, w1 = *(const float4*)(vp + 4);
      vstage[seg*8+0][key] = (_Float16)w0.x;
      vstage[seg*8+1][key] = (_Float16)w0.y;
      vstage[seg*8+2][key] = (_Float16)w0.z;
      vstage[seg*8+3][key] = (_Float16)w0.w;
      vstage[seg*8+4][key] = (_Float16)w1.x;
      vstage[seg*8+5][key] = (_Float16)w1.y;
      vstage[seg*8+6][key] = (_Float16)w1.z;
      vstage[seg*8+7][key] = (_Float16)w1.w;

      if (kb + 32 < S_) {                       // CDNA5 global_prefetch_b8
        __builtin_prefetch(kp + 32 * D_, 0, 1);
        __builtin_prefetch(vp + 32 * D_, 0, 1);
      }
    }
    __syncthreads();

    // ---- scores: two 16x16 blocks, accumulate over dk = 0..63 (2 WMMAs each) ----
    v8f s0 = {}, s1 = {};
    s0 = wmma_f16(qa0, ldB_lds(&kstage[0][0],   64, lane), s0);
    s0 = wmma_f16(qa1, ldB_lds(&kstage[0][32],  64, lane), s0);
    s1 = wmma_f16(qa0, ldB_lds(&kstage[16][0],  64, lane), s1);
    s1 = wmma_f16(qa1, ldB_lds(&kstage[16][32], 64, lane), s1);

    const float km0 = mk[kb + m0];
    const float km1 = mk[kb + 16 + m0];

    // ---- online softmax (row = r + 8*lh, col = lane%16 within each half) ----
    float nm[8];
#pragma unroll
    for (int r = 0; r < 8; ++r) {
      float x0 = (km0 != 0.f) ? s0[r] * 0.125f : -1e9f;  // 1/sqrt(64)
      float x1 = (km1 != 0.f) ? s1[r] * 0.125f : -1e9f;
      s0[r] = x0; s1[r] = x1;
      float rm = fmaxf(x0, x1);
      rm = fmaxf(rm, __shfl_xor(rm, 1));
      rm = fmaxf(rm, __shfl_xor(rm, 2));
      rm = fmaxf(rm, __shfl_xor(rm, 4));
      rm = fmaxf(rm, __shfl_xor(rm, 8));                 // reduce within 16-lane half
      nm[r] = fmaxf(mrow[r], rm);
    }
    _Float16* pp = &pstage[wave][0][0];
#pragma unroll
    for (int r = 0; r < 8; ++r) {
      float corr = __expf(mrow[r] - nm[r]);
      float p0 = __expf(s0[r] - nm[r]) * km0;
      float p1 = __expf(s1[r] - nm[r]) * km1;
      float rs = p0 + p1;
      rs += __shfl_xor(rs, 1); rs += __shfl_xor(rs, 2);
      rs += __shfl_xor(rs, 4); rs += __shfl_xor(rs, 8);
      lrow[r] = lrow[r] * corr + rs;
      mrow[r] = nm[r];
      o0[r] *= corr; o1[r] *= corr; o2[r] *= corr; o3[r] *= corr;
      pp[(r + 8 * lh) * 32 + m0]      = (_Float16)p0;    // D-layout -> row-major LDS
      pp[(r + 8 * lh) * 32 + 16 + m0] = (_Float16)p1;
    }

    // ---- P*V: read P back in A-layout (same-wave DS ops are in order) ----
    v16h pa = ldA_lds(pp, 32, lane);
    o0 = wmma_f16(pa, ldB_lds(&vstage[0][0],  32, lane), o0);
    o1 = wmma_f16(pa, ldB_lds(&vstage[16][0], 32, lane), o1);
    o2 = wmma_f16(pa, ldB_lds(&vstage[32][0], 32, lane), o2);
    o3 = wmma_f16(pa, ldB_lds(&vstage[48][0], 32, lane), o3);
    __syncthreads();
  }

  // ---- normalize and write x (B,S,D) ----
  float* Xb = X + (bs + qbase) * D_ + hd * DK_;
#pragma unroll
  for (int r = 0; r < 8; ++r) {
    const int row = r + 8 * lh;
    const float inv = (lrow[r] > 0.f) ? 1.f / lrow[r] : 0.f;
    float* xr = Xb + (size_t)row * D_;
    xr[ 0 + m0] = o0[r] * inv;
    xr[16 + m0] = o1[r] * inv;
    xr[32 + m0] = o2[r] * inv;
    xr[48 + m0] = o3[r] * inv;
  }
}

// =====================================================================
// Kernel 2: out = (X @ Wp^T + bp) * mask_row.  Wave -> 16x64 output tile.
// =====================================================================
__global__ __launch_bounds__(256) void proj_gemm_kernel(
    const float* __restrict__ X, const float* __restrict__ Wp,
    const float* __restrict__ bp, const float* __restrict__ mask,
    float* __restrict__ out) {
  const int tid = threadIdx.x, wave = tid >> 5, lane = tid & 31;
  const int m0 = lane & 15, lh = (lane >> 4) & 1;
  const int mbase = blockIdx.x * 16;                  // rows in [0, B*S)
  const int nbase = blockIdx.y * 512 + wave * 64;     // output cols

  v8f a0 = {}, a1 = {}, a2 = {}, a3 = {};
  const float* Xb = X + (size_t)mbase * D_;
  for (int kk = 0; kk < D_; kk += 32) {
    v16h xa = ldA_f32(Xb + kk, D_, lane);
    // B[k][n] = Wp[n][k]: Wp rows are N-major with k contiguous -> B loader fits
    a0 = wmma_f16(xa, ldB_f32(Wp + (size_t)(nbase +  0) * D_ + kk, D_, lane), a0);
    a1 = wmma_f16(xa, ldB_f32(Wp + (size_t)(nbase + 16) * D_ + kk, D_, lane), a1);
    a2 = wmma_f16(xa, ldB_f32(Wp + (size_t)(nbase + 32) * D_ + kk, D_, lane), a2);
    a3 = wmma_f16(xa, ldB_f32(Wp + (size_t)(nbase + 48) * D_ + kk, D_, lane), a3);
  }
  const float bv0 = bp[nbase +  0 + m0];
  const float bv1 = bp[nbase + 16 + m0];
  const float bv2 = bp[nbase + 32 + m0];
  const float bv3 = bp[nbase + 48 + m0];
#pragma unroll
  for (int r = 0; r < 8; ++r) {
    const int row = mbase + r + 8 * lh;
    const float mv = mask[row];                       // flat (B*S)
    float* op = out + (size_t)row * OUT_;
    op[nbase +  0 + m0] = (a0[r] + bv0) * mv;
    op[nbase + 16 + m0] = (a1[r] + bv1) * mv;
    op[nbase + 32 + m0] = (a2[r] + bv2) * mv;
    op[nbase + 48 + m0] = (a3[r] + bv3) * mv;
  }
}

extern "C" void kernel_launch(void* const* d_in, const int* in_sizes, int n_in,
                              void* d_out, int out_size, void* d_ws, size_t ws_size,
                              hipStream_t stream) {
  (void)in_sizes; (void)n_in; (void)out_size; (void)ws_size;
  const float* Q    = (const float*)d_in[0];
  const float* K    = (const float*)d_in[1];
  const float* V    = (const float*)d_in[2];
  const float* mask = (const float*)d_in[3];
  const float* Wp   = (const float*)d_in[4];
  const float* bp   = (const float*)d_in[5];
  float* out = (float*)d_out;
  float* X   = (float*)d_ws;   // B*S*D f32 = 16 MB scratch for attention output

  attn_flash_kernel<<<dim3(B_ * H_ * (S_ / 128)), dim3(256), 0, stream>>>(
      Q, K, V, mask, X);
  proj_gemm_kernel<<<dim3((B_ * S_) / 16, OUT_ / 512), dim3(256), 0, stream>>>(
      X, Wp, bp, mask, out);
}
